// HardAttention_76261439307988
// MI455X (gfx1250) — matrix operand order repaired and checked by reading, not compiled, not run on hardware
//
#include <hip/hip_runtime.h>
#include <math.h>

// Problem constants from the reference: B=32, T=4096, H=256.
#define B_ 32
#define T_ 4096
#define H_ 256

typedef __attribute__((ext_vector_type(16))) __bf16 v16bf;
typedef __attribute__((ext_vector_type(8)))  float  v8f;
typedef __attribute__((ext_vector_type(4)))  unsigned int u32x4;

// Number of 16-byte chunks in the packed W2 image: 128 frags * 64 chunks.
#define W2P_CHUNKS (128 * 64)   // = 8192 chunks = 128 KB

// Pack 16 f32 (four float4, in order) into a v16bf fragment (bf16 convert).
__device__ __forceinline__ v16bf pack16(float4 a, float4 b, float4 c, float4 d) {
    v16bf r;
    r[0]  = (__bf16)a.x; r[1]  = (__bf16)a.y; r[2]  = (__bf16)a.z; r[3]  = (__bf16)a.w;
    r[4]  = (__bf16)b.x; r[5]  = (__bf16)b.y; r[6]  = (__bf16)b.z; r[7]  = (__bf16)b.w;
    r[8]  = (__bf16)c.x; r[9]  = (__bf16)c.y; r[10] = (__bf16)c.z; r[11] = (__bf16)c.w;
    r[12] = (__bf16)d.x; r[13] = (__bf16)d.y; r[14] = (__bf16)d.z; r[15] = (__bf16)d.w;
    return r;
}

// ---------------------------------------------------------------------------
// Kernel 0a: hb[b][n] = attn_b[n] + sum_k hidden[b][k] * attn_w[n][k]
// ---------------------------------------------------------------------------
__global__ void __launch_bounds__(256) hb_kernel(const float* __restrict__ hidden,
                                                 const float* __restrict__ attnw,
                                                 const float* __restrict__ attnb,
                                                 float* __restrict__ hb) {
    const int b = blockIdx.x;
    const int n = threadIdx.x;
    const float* __restrict__ hrow = hidden + (size_t)b * H_;
    const float* __restrict__ wrow = attnw + (size_t)n * (2 * H_);   // W1 = attn_w[:, :H]
    float s = attnb[n];
    #pragma unroll 4
    for (int k = 0; k < H_; k += 4) {
        float4 hv = *(const float4*)(hrow + k);
        float4 wv = *(const float4*)(wrow + k);
        s += hv.x * wv.x + hv.y * wv.y + hv.z * wv.z + hv.w * wv.w;
    }
    hb[(size_t)b * H_ + n] = s;
}

// ---------------------------------------------------------------------------
// Kernel 0b: pre-pack W2 = attn_w[:, H:] into bf16 in the wave32 WMMA
// B-fragment order used here (contiguous k, as A). Chunk c (16 B) encodes: frag f = c>>6 (f = nc*8+kc),
// r = c&63, plane = r>>5 (elements 0..7 vs 8..15), lane l = r&31.
// B 32x16 bf16 order used here (same k order as A): column N = nc*16 + (l&15);
// K = kc*32 + (l>>4)*16 + plane*8 + (0..7), ascending packed order.
// ---------------------------------------------------------------------------
__global__ void __launch_bounds__(256) packw2_kernel(const float* __restrict__ attnw,
                                                     u32x4* __restrict__ w2p) {
    const int c = blockIdx.x * 256 + threadIdx.x;   // 0 .. 8191
    const int f     = c >> 6;
    const int r     = c & 63;
    const int plane = r >> 5;
    const int l     = r & 31;
    const int nc    = f >> 3;
    const int kc    = f & 7;
    const int N     = nc * 16 + (l & 15);
    const int Kb    = kc * 32 + (l >> 4) * 16 + plane * 8;
    const float* src = attnw + (size_t)N * (2 * H_) + H_ + Kb;
    float4 a = *(const float4*)(src);
    float4 b = *(const float4*)(src + 4);
    union { __bf16 h[8]; u32x4 q; } u;
    u.h[0] = (__bf16)a.x; u.h[1] = (__bf16)a.y; u.h[2] = (__bf16)a.z; u.h[3] = (__bf16)a.w;
    u.h[4] = (__bf16)b.x; u.h[5] = (__bf16)b.y; u.h[6] = (__bf16)b.z; u.h[7] = (__bf16)b.w;
    w2p[c] = u.q;
}

// ---------------------------------------------------------------------------
// Kernel 1: fused  scores[b][t] = v . relu(enc_row @ W2^T + hb[b])
// One wave32 per 16-row M-tile. The 8 waves of a block share a 128 KB LDS
// copy of the pre-packed bf16 W2 (CDNA5 WGP has 320 KB LDS).
// Hot loop: 2x ds_load_b128 -> v_wmma_f32_16x16x32_bf16, no VALU cvts.
// ---------------------------------------------------------------------------
__global__ void __launch_bounds__(256) score_kernel(const float* __restrict__ enc,
                                                    const u32x4* __restrict__ w2p,
                                                    const float* __restrict__ hb,
                                                    const float* __restrict__ vvec,
                                                    float* __restrict__ scores) {
    __shared__ u32x4 w2lds[W2P_CHUNKS];   // 128 KB

    const int tid  = threadIdx.x;
    const int lane = tid & 31;
    const int wave = tid >> 5;
    const int hi   = lane >> 4;     // 16-lane half (0/1)
    const int lrow = lane & 15;
    const int m0   = (blockIdx.x * 8 + wave) * 16;   // first enc row of this tile

    // --- Cooperative copy of packed W2 into LDS (uniform control flow) ------
    #pragma unroll 4
    for (int i = tid; i < W2P_CHUNKS; i += 256)
        w2lds[i] = w2p[i];

    // --- Preload all 8 A-fragments (this wave's 16 enc rows, K=0..255) ------
    // A 16x32 bf16 order used here (same k order as B): lanes 0-15 hold row lrow, K = k0+0..15;
    // lanes 16-31 hold row lrow, K = k0+16..31.
    v16bf afrag[8];
    {
        const float* __restrict__ rowp = enc + (size_t)(m0 + lrow) * H_;
        const int ks = hi * 16;
        #pragma unroll
        for (int kc = 0; kc < 8; ++kc) {
            const float* p = rowp + kc * 32 + ks;
            float4 f0 = *(const float4*)(p);
            float4 f1 = *(const float4*)(p + 4);
            float4 f2 = *(const float4*)(p + 8);
            float4 f3 = *(const float4*)(p + 12);
            afrag[kc] = pack16(f0, f1, f2, f3);
        }
    }

    __syncthreads();

    float acc[8];
    #pragma unroll
    for (int i = 0; i < 8; ++i) acc[i] = 0.0f;

    // --- Loop over 16 N-chunks of 16 columns each ---------------------------
    for (int nc = 0; nc < 16; ++nc) {
        const int ncol = nc * 16 + lrow;
        v8f c = {};
        #pragma unroll
        for (int kc = 0; kc < 8; ++kc) {
            const int f = nc * 8 + kc;
            union { struct { u32x4 lo, hi; } s; v16bf v; } bu;
            bu.s.lo = w2lds[f * 64 + lane];        // elements 0..7  (plane 0)
            bu.s.hi = w2lds[f * 64 + 32 + lane];   // elements 8..15 (plane 1)
            // D = A*B + C, f32 accumulate
            c = __builtin_amdgcn_wmma_f32_16x16x32_bf16(
                    false, afrag[kc], false, bu.v, (short)0, c, false, false);
        }
        // C layout: VGPR i, lanes 0-15 -> (M=i, N=ncol); lanes 16-31 -> (M=i+8).
        const float vn = vvec[ncol];
        #pragma unroll
        for (int i = 0; i < 8; ++i) {
            const int r  = m0 + hi * 8 + i;          // enc flat row = t*B + b
            const float e = c[i] + hb[(size_t)(r & (B_ - 1)) * H_ + ncol];
            acc[i] += fmaxf(e, 0.0f) * vn;
        }
    }

    // --- Reduce each row's partial score across its 16-lane half ------------
    #pragma unroll
    for (int i = 0; i < 8; ++i) {
        float s = acc[i];
        s += __shfl_xor(s, 1, 32);
        s += __shfl_xor(s, 2, 32);
        s += __shfl_xor(s, 4, 32);
        s += __shfl_xor(s, 8, 32);
        acc[i] = s;
    }
    if (lrow == 0) {   // lane 0 writes rows 0..7, lane 16 writes rows 8..15
        #pragma unroll
        for (int i = 0; i < 8; ++i) {
            const int r = m0 + hi * 8 + i;
            const int b = r & (B_ - 1);
            const int t = r >> 5;
            scores[(size_t)b * T_ + t] = acc[i];
        }
    }
}

// ---------------------------------------------------------------------------
// Kernel 2: in-place softmax over T per batch row (out is [B, 1, T]).
// ---------------------------------------------------------------------------
__global__ void __launch_bounds__(256) softmax_kernel(float* __restrict__ out) {
    __shared__ float red[256];
    const int b   = blockIdx.x;
    const int tid = threadIdx.x;
    float* row = out + (size_t)b * T_;

    float sv[T_ / 256];
    float m = -INFINITY;
    #pragma unroll
    for (int j = 0; j < T_ / 256; ++j) {
        sv[j] = row[j * 256 + tid];
        m = fmaxf(m, sv[j]);
    }
    red[tid] = m;
    __syncthreads();
    for (int s = 128; s > 0; s >>= 1) {
        if (tid < s) red[tid] = fmaxf(red[tid], red[tid + s]);
        __syncthreads();
    }
    m = red[0];
    __syncthreads();

    float sum = 0.0f;
    #pragma unroll
    for (int j = 0; j < T_ / 256; ++j) {
        sv[j] = __expf(sv[j] - m);
        sum += sv[j];
    }
    red[tid] = sum;
    __syncthreads();
    for (int s = 128; s > 0; s >>= 1) {
        if (tid < s) red[tid] += red[tid + s];
        __syncthreads();
    }
    const float inv = 1.0f / red[0];
    #pragma unroll
    for (int j = 0; j < T_ / 256; ++j)
        row[j * 256 + tid] = sv[j] * inv;
}

// ---------------------------------------------------------------------------
extern "C" void kernel_launch(void* const* d_in, const int* in_sizes, int n_in,
                              void* d_out, int out_size, void* d_ws, size_t ws_size,
                              hipStream_t stream) {
    const float* hidden = (const float*)d_in[0];  // [B, H]
    const float* enc    = (const float*)d_in[1];  // [T, B, H]
    const float* attnw  = (const float*)d_in[2];  // [H, 2H]
    const float* attnb  = (const float*)d_in[3];  // [H]
    const float* vvec   = (const float*)d_in[4];  // [H]
    float*       out    = (float*)d_out;          // [B, 1, T] == B*T floats

    // Workspace layout: hb [32 KB] | packed bf16 W2 [128 KB]
    float* hb  = (float*)d_ws;
    u32x4* w2p = (u32x4*)((char*)d_ws + B_ * H_ * sizeof(float));

    hb_kernel<<<B_, 256, 0, stream>>>(hidden, attnw, attnb, hb);
    packw2_kernel<<<W2P_CHUNKS / 256, 256, 0, stream>>>(attnw, w2p);

    // 131072 rows / 16 per wave = 8192 tiles; 8 waves (256 thr) per block.
    score_kernel<<<(B_ * T_ / 16) / 8, 256, 0, stream>>>(enc, w2p, hb, vvec, out);

    softmax_kernel<<<B_, 256, 0, stream>>>(out);
}
